// MultiHeadAttention_9947144257796
// MI455X (gfx1250) — matrix operand order repaired and checked by reading, not compiled, not run on hardware
//
#include <hip/hip_runtime.h>
#include <cmath>

typedef __bf16 bf16_t;
typedef __attribute__((ext_vector_type(16))) __bf16 v16bf;
typedef __attribute__((ext_vector_type(8)))  float  v8f;
typedef __attribute__((ext_vector_type(4)))  float  v4f;   // native vector for NT stores

#define TW 40     // padded LDS row stride (bf16) for 32-deep GEMM tiles
#define QROW 72   // padded LDS row stride (bf16) for 64-wide attention tiles (144B = 16B-aligned)
#define SROW 1028 // padded LDS row stride (f32) for the 1024-wide score strip

// ---------------- WMMA helpers (ISA 7.12.2 layouts) ----------------
static __device__ __forceinline__ v8f wmma_bf16(v16bf a, v16bf b, v8f c) {
  return __builtin_amdgcn_wmma_f32_16x16x32_bf16(false, a, false, b, (short)0, c, false, false);
}

// A: 16x32 bf16 from row-major LDS tile As[row][k].
// lanes 0-15: M=lane, elems 0-7 = K0..7, elems 8-15 = K16..23
// lanes 16-31: M=lane-16, elems 0-7 = K8..15, elems 8-15 = K24..31
static __device__ __forceinline__ v16bf fragA(const bf16_t* As, int ld, int rowOff, int kOff) {
  const int lane = threadIdx.x & 31;
  const bf16_t* p = As + (rowOff + (lane & 15)) * ld + kOff + ((lane & 16) ? 8 : 0);
  v16bf a;
#pragma unroll
  for (int i = 0; i < 8; ++i) a[i] = p[i];
#pragma unroll
  for (int i = 0; i < 8; ++i) a[8 + i] = p[16 + i];
  return a;
}

// Same A fragment but gathered from an f32 LDS tile with on-the-fly bf16 convert.
static __device__ __forceinline__ v16bf fragA_f32(const float* S, int ld, int rowOff, int kOff) {
  const int lane = threadIdx.x & 31;
  const float* p = S + (size_t)(rowOff + (lane & 15)) * ld + kOff + ((lane & 16) ? 8 : 0);
  v16bf a;
#pragma unroll
  for (int i = 0; i < 8; ++i) a[i] = (bf16_t)p[i];
#pragma unroll
  for (int i = 0; i < 8; ++i) a[8 + i] = (bf16_t)p[16 + i];
  return a;
}

// B: 32x16 bf16, stored N-major in LDS: BsT[n][k].
// lanes 0-15: N=lane, elems 0-7 = K0..7, elems 8-15 = K16..23 ; lanes 16-31: N=lane-16, elems 0-7 = K8..15, elems 8-15 = K24..31
static __device__ __forceinline__ v16bf fragB(const bf16_t* BsT, int ld, int colOff, int kOff) {
  const int lane = threadIdx.x & 31;
  const bf16_t* p = BsT + (colOff + (lane & 15)) * ld + kOff + ((lane & 16) ? 8 : 0);
  v16bf b;
#pragma unroll
  for (int i = 0; i < 8; ++i) b[i] = p[i];
#pragma unroll
  for (int i = 0; i < 8; ++i) b[8 + i] = p[16 + i];
  return b;
}

// C/D 16x16 f32: lanes 0-15: N=lane, VGPR r -> M=r ; lanes 16-31: N=lane-16, M=8+r
static __device__ __forceinline__ void storeC_bf16(bf16_t* C, int ldc, v8f c) {
  const int lane = threadIdx.x & 31;
  const int n = lane & 15;
  const int mb = (lane & 16) ? 8 : 0;
#pragma unroll
  for (int r = 0; r < 8; ++r) C[(mb + r) * ldc + n] = (bf16_t)c[r];
}

// ---------------- LDS staging for 128-thread GEMM blocks ----------------
static __device__ __forceinline__ void stage_f32_A(bf16_t* dst, const float* src, int ld, int tid) {
#pragma unroll
  for (int i = 0; i < 4; ++i) {
    int lin = (tid + i * 128) * 4;                 // 64x32 tile, 512 float4s
    int r = lin >> 5, c = lin & 31;
    float4 v = *(const float4*)(src + (size_t)r * ld + c);
    bf16_t* d = dst + r * TW + c;
    d[0] = (bf16_t)v.x; d[1] = (bf16_t)v.y; d[2] = (bf16_t)v.z; d[3] = (bf16_t)v.w;
  }
}
static __device__ __forceinline__ void stage_f32_BT(bf16_t* dst, const float* src, int ld, int tid) {
#pragma unroll
  for (int i = 0; i < 4; ++i) {
    int lin = (tid + i * 128) * 4;                 // 32x64 tile -> [n][k]
    int k = lin >> 6, n = lin & 63;
    float4 v = *(const float4*)(src + (size_t)k * ld + n);
    dst[(n + 0) * TW + k] = (bf16_t)v.x;
    dst[(n + 1) * TW + k] = (bf16_t)v.y;
    dst[(n + 2) * TW + k] = (bf16_t)v.z;
    dst[(n + 3) * TW + k] = (bf16_t)v.w;
  }
}
static __device__ __forceinline__ void stage_bf16_A(bf16_t* dst, const bf16_t* src, int ld, int tid) {
#pragma unroll
  for (int i = 0; i < 2; ++i) {
    int lin = (tid + i * 128) * 8;                 // 64x32 tile, 16B copies
    int r = lin >> 5, c = lin & 31;
    *(uint4*)(dst + r * TW + c) = *(const uint4*)(src + (size_t)r * ld + c);
  }
}

// ---------------- 1) per-head input projection ----------------
// Out[m, h*64+e] = sum_d X[m,d] * W[h,d,e]   (M=8192, N=1024, K=1024)
__global__ __launch_bounds__(128) void proj_kernel(const float* __restrict__ X,
                                                   const float* __restrict__ W,
                                                   bf16_t* __restrict__ Out) {
  __shared__ bf16_t As[64 * TW];
  __shared__ bf16_t Bs[64 * TW];
  const int tid = threadIdx.x;
  const int m0 = blockIdx.x * 64;
  const int h  = blockIdx.y;                 // 64 output cols per head
  const int wave = tid >> 5;
  const int wm = (wave >> 1) * 32, wn = (wave & 1) * 32;
  v8f c00 = {}, c01 = {}, c10 = {}, c11 = {};
  for (int kk = 0; kk < 1024; kk += 32) {
    stage_f32_A(As, X + (size_t)m0 * 1024 + kk, 1024, tid);
    stage_f32_BT(Bs, W + (size_t)h * 65536 + (size_t)kk * 64, 64, tid);
    __syncthreads();
    v16bf a0 = fragA(As, TW, wm, 0), a1 = fragA(As, TW, wm + 16, 0);
    v16bf b0 = fragB(Bs, TW, wn, 0), b1 = fragB(Bs, TW, wn + 16, 0);
    c00 = wmma_bf16(a0, b0, c00);
    c01 = wmma_bf16(a0, b1, c01);
    c10 = wmma_bf16(a1, b0, c10);
    c11 = wmma_bf16(a1, b1, c11);
    __syncthreads();
  }
  bf16_t* C = Out + (size_t)m0 * 1024 + h * 64;
  storeC_bf16(C + (wm +  0) * 1024 + wn,      1024, c00);
  storeC_bf16(C + (wm +  0) * 1024 + wn + 16, 1024, c01);
  storeC_bf16(C + (wm + 16) * 1024 + wn,      1024, c10);
  storeC_bf16(C + (wm + 16) * 1024 + wn + 16, 1024, c11);
}

// ---------------- 2) fused scores + softmax + AV ----------------
// Per (hb, 64-row q-block): score strip lives in LDS; attn hits HBM exactly once (NT stores).
__global__ __launch_bounds__(256) void fused_attn_kernel(const bf16_t* __restrict__ Qb,
                                                         const bf16_t* __restrict__ Kb,
                                                         const bf16_t* __restrict__ Vb,
                                                         float* __restrict__ attn,
                                                         bf16_t* __restrict__ Ob) {
  extern __shared__ char smem[];
  float*  Sf = (float*)smem;                                       // 64 x SROW f32 (257KB)
  bf16_t* Qs = (bf16_t*)(smem + (size_t)64 * SROW * 4);            // 64 x QROW bf16
  bf16_t* Ks = (bf16_t*)(smem + (size_t)64 * SROW * 4 + 64 * QROW * 2);
  bf16_t* Vs = Ks;                                                 // reused in phase 3

  const int tid  = threadIdx.x;
  const int lane = tid & 31;
  const int wave = tid >> 5;
  const int q0 = blockIdx.x * 64;
  const int hb = blockIdx.y;                 // h*8 + b
  const int h = hb >> 3, b = hb & 7;

  const int rt = wave >> 1;                  // row-tile (16 rows each), 0..3
  const int c0 = (wave & 1) * 2;             // first of two 16-wide col-tiles

  // stage Q tile 64x64 once
  const bf16_t* Qsrc = Qb + (size_t)(b * 1024 + q0) * 1024 + h * 64;
#pragma unroll
  for (int i = 0; i < 2; ++i) {
    int lin = (tid + i * 256) * 8;
    int r = lin >> 6, c = lin & 63;
    *(uint4*)(Qs + r * QROW + c) = *(const uint4*)(Qsrc + (size_t)r * 1024 + c);
  }

  // ---- phase 1: scores (scaled) into Sf ----
  for (int kc = 0; kc < 16; ++kc) {
    const bf16_t* Ksrc = Kb + (size_t)(b * 1024 + kc * 64) * 1024 + h * 64;
    __syncthreads();                         // previous Ks consumers done
#pragma unroll
    for (int i = 0; i < 2; ++i) {
      int lin = (tid + i * 256) * 8;
      int r = lin >> 6, c = lin & 63;        // Ks[kcol][e] — already N-major
      *(uint4*)(Ks + r * QROW + c) = *(const uint4*)(Ksrc + (size_t)r * 1024 + c);
    }
    __syncthreads();
    v8f a0 = {}, a1 = {};
#pragma unroll
    for (int e0 = 0; e0 < 64; e0 += 32) {
      v16bf qa  = fragA(Qs, QROW, rt * 16, e0);
      v16bf kb0 = fragB(Ks, QROW, c0 * 16, e0);
      v16bf kb1 = fragB(Ks, QROW, c0 * 16 + 16, e0);
      a0 = wmma_bf16(qa, kb0, a0);
      a1 = wmma_bf16(qa, kb1, a1);
    }
    const float sc = 0.03125f;               // 1/sqrt(1024)
    const int nn = lane & 15;
    const int mb = (lane & 16) ? 8 : 0;
    float* Sp = Sf + (size_t)(rt * 16 + mb) * SROW + kc * 64 + c0 * 16 + nn;
#pragma unroll
    for (int r = 0; r < 8; ++r) {
      Sp[(size_t)r * SROW]      = a0[r] * sc;
      Sp[(size_t)r * SROW + 16] = a1[r] * sc;
    }
  }
  __syncthreads();

  // ---- phase 2: softmax per row; probs back to Sf + NT-stream to global attn ----
  {
    float* outbase = attn + ((size_t)hb << 20) + (size_t)q0 * 1024;
#pragma unroll
    for (int j = 0; j < 8; ++j) {
      const int row = wave * 8 + j;
      float* p = Sf + (size_t)row * SROW;
      v4f v[8];
      float mx = -INFINITY;
#pragma unroll
      for (int i = 0; i < 8; ++i) {
        v[i] = *(const v4f*)(p + i * 128 + lane * 4);
        mx = fmaxf(mx, fmaxf(fmaxf(v[i].x, v[i].y), fmaxf(v[i].z, v[i].w)));
      }
#pragma unroll
      for (int off = 16; off > 0; off >>= 1) mx = fmaxf(mx, __shfl_xor(mx, off, 32));
      float sum = 0.f;
#pragma unroll
      for (int i = 0; i < 8; ++i) {
        v[i].x = __expf(v[i].x - mx); v[i].y = __expf(v[i].y - mx);
        v[i].z = __expf(v[i].z - mx); v[i].w = __expf(v[i].w - mx);
        sum += v[i].x + v[i].y + v[i].z + v[i].w;
      }
#pragma unroll
      for (int off = 16; off > 0; off >>= 1) sum += __shfl_xor(sum, off, 32);
      const float inv = 1.f / sum;
      float* g = outbase + (size_t)row * 1024;
#pragma unroll
      for (int i = 0; i < 8; ++i) {
        v[i].x *= inv; v[i].y *= inv; v[i].z *= inv; v[i].w *= inv;
        *(v4f*)(p + i * 128 + lane * 4) = v[i];
        __builtin_nontemporal_store(v[i], (v4f*)(g + i * 128 + lane * 4));
      }
    }
  }

  // ---- phase 3: out = P @ V (P read from Sf with f32->bf16 gather) ----
  v8f o0 = {}, o1 = {};
  for (int kk = 0; kk < 1024; kk += 64) {
    const bf16_t* Vsrc = Vb + (size_t)(b * 1024 + kk) * 1024 + h * 64;
    __syncthreads();                         // Vs free
#pragma unroll
    for (int i = 0; i < 2; ++i) {
      int lin = (tid + i * 256) * 8;
      int k = lin >> 6, e = lin & 63;        // Vs[e][k] — transposed to N-major
      uint4 vv = *(const uint4*)(Vsrc + (size_t)k * 1024 + e);
      const bf16_t* pp = (const bf16_t*)&vv;
#pragma unroll
      for (int j = 0; j < 8; ++j) Vs[(e + j) * QROW + k] = pp[j];
    }
    __syncthreads();
#pragma unroll
    for (int ks = 0; ks < 64; ks += 32) {
      v16bf pa  = fragA_f32(Sf, SROW, rt * 16, kk + ks);
      v16bf vb0 = fragB(Vs, QROW, c0 * 16, ks);
      v16bf vb1 = fragB(Vs, QROW, c0 * 16 + 16, ks);
      o0 = wmma_bf16(pa, vb0, o0);
      o1 = wmma_bf16(pa, vb1, o1);
    }
  }
  bf16_t* C = Ob + (size_t)(b * 1024 + q0) * 1024 + h * 64;   // head-major concat layout
  storeC_bf16(C + (rt * 16) * 1024 + c0 * 16,      1024, o0);
  storeC_bf16(C + (rt * 16) * 1024 + c0 * 16 + 16, 1024, o1);
}

// ---------------- 3) out proj + bias + residual ----------------
__global__ __launch_bounds__(128) void outproj_kernel(const bf16_t* __restrict__ Ob,
                                                      const float* __restrict__ Wp,
                                                      const float* __restrict__ bp,
                                                      const float* __restrict__ query,
                                                      float* __restrict__ Ypre) {
  __shared__ bf16_t As[64 * TW];
  __shared__ bf16_t Bs[64 * TW];
  const int tid = threadIdx.x;
  const int m0 = blockIdx.x * 64;
  const int n0 = blockIdx.y * 64;
  const int wave = tid >> 5;
  const int wm = (wave >> 1) * 32, wn = (wave & 1) * 32;
  v8f c00 = {}, c01 = {}, c10 = {}, c11 = {};
  for (int kk = 0; kk < 1024; kk += 32) {
    stage_bf16_A(As, Ob + (size_t)m0 * 1024 + kk, 1024, tid);
    stage_f32_BT(Bs, Wp + (size_t)kk * 1024 + n0, 1024, tid);
    __syncthreads();
    v16bf a0 = fragA(As, TW, wm, 0), a1 = fragA(As, TW, wm + 16, 0);
    v16bf b0 = fragB(Bs, TW, wn, 0), b1 = fragB(Bs, TW, wn + 16, 0);
    c00 = wmma_bf16(a0, b0, c00);
    c01 = wmma_bf16(a0, b1, c01);
    c10 = wmma_bf16(a1, b0, c10);
    c11 = wmma_bf16(a1, b1, c11);
    __syncthreads();
  }
  const int lane = tid & 31;
  const int nn = lane & 15;
  const int mb = (lane & 16) ? 8 : 0;
  v8f acc[4] = {c00, c01, c10, c11};
  const int mo[4] = {wm, wm, wm + 16, wm + 16};
  const int no[4] = {wn, wn + 16, wn, wn + 16};
#pragma unroll
  for (int t = 0; t < 4; ++t) {
#pragma unroll
    for (int r = 0; r < 8; ++r) {
      const int m = m0 + mo[t] + mb + r;
      const int n = n0 + no[t] + nn;
      Ypre[(size_t)m * 1024 + n] = acc[t][r] + bp[n] + query[(size_t)m * 1024 + n];
    }
  }
}

// ---------------- 4) LayerNorm (wave32 per row) ----------------
__global__ __launch_bounds__(256) void ln_kernel(const float* __restrict__ Ypre,
                                                 const float* __restrict__ gamma,
                                                 const float* __restrict__ beta,
                                                 float* __restrict__ y) {
  const int lane = threadIdx.x & 31;
  const int row  = blockIdx.x * 8 + (threadIdx.x >> 5);
  const float* p = Ypre + (size_t)row * 1024;
  float4 v[8];
  float s = 0.f, s2 = 0.f;
#pragma unroll
  for (int i = 0; i < 8; ++i) {
    v[i] = *(const float4*)(p + i * 128 + lane * 4);
    s  += v[i].x + v[i].y + v[i].z + v[i].w;
    s2 += v[i].x * v[i].x + v[i].y * v[i].y + v[i].z * v[i].z + v[i].w * v[i].w;
  }
#pragma unroll
  for (int off = 16; off > 0; off >>= 1) {
    s  += __shfl_xor(s, off, 32);
    s2 += __shfl_xor(s2, off, 32);
  }
  const float mu  = s * (1.f / 1024.f);
  const float var = s2 * (1.f / 1024.f) - mu * mu;
  const float inv = rsqrtf(var + 1e-5f);
  float* o = y + (size_t)row * 1024;
#pragma unroll
  for (int i = 0; i < 8; ++i) {
    const int c = i * 128 + lane * 4;
    float4 g  = *(const float4*)(gamma + c);
    float4 be = *(const float4*)(beta + c);
    float4 r;
    r.x = (v[i].x - mu) * inv * g.x + be.x;
    r.y = (v[i].y - mu) * inv * g.y + be.y;
    r.z = (v[i].z - mu) * inv * g.z + be.z;
    r.w = (v[i].w - mu) * inv * g.w + be.w;
    *(float4*)(o + c) = r;
  }
}

// ---------------- host-side launcher ----------------
extern "C" void kernel_launch(void* const* d_in, const int* in_sizes, int n_in,
                              void* d_out, int out_size, void* d_ws, size_t ws_size,
                              hipStream_t stream) {
  (void)in_sizes; (void)n_in; (void)out_size; (void)ws_size;
  const float* query  = (const float*)d_in[0];   // [8,1024,1024]
  const float* keys   = (const float*)d_in[1];
  const float* values = (const float*)d_in[2];
  const float* Wq     = (const float*)d_in[3];   // [16,1024,64]
  const float* Wk     = (const float*)d_in[4];
  const float* Wv     = (const float*)d_in[5];
  const float* Wp     = (const float*)d_in[6];   // [1024,1024]
  const float* bp     = (const float*)d_in[7];
  const float* gamma  = (const float*)d_in[8];
  const float* beta   = (const float*)d_in[9];

  float* y    = (float*)d_out;                                // output 0: [8,1024,1024]
  float* attn = (float*)d_out + (size_t)8 * 1024 * 1024;      // output 1: [128,1024,1024]

  char* ws = (char*)d_ws;
  bf16_t* Qb = (bf16_t*)(ws);                        // 16 MB  [8192,1024] bf16
  bf16_t* Kb = (bf16_t*)(ws + (size_t)(16u << 20));
  bf16_t* Vb = (bf16_t*)(ws + (size_t)(32u << 20));
  bf16_t* Ob = (bf16_t*)(ws + (size_t)(48u << 20));
  float*  Ypre = (float*)(ws);                       // 32 MB, reuses Qb+Kb (dead by then)

  const dim3 blk(128);
  proj_kernel<<<dim3(128, 16), blk, 0, stream>>>(query,  Wq, Qb);
  proj_kernel<<<dim3(128, 16), blk, 0, stream>>>(keys,   Wk, Kb);
  proj_kernel<<<dim3(128, 16), blk, 0, stream>>>(values, Wv, Vb);

  const size_t smem = (size_t)64 * SROW * 4 + 2u * 64 * QROW * 2;   // 281,600 B < 320 KB
  fused_attn_kernel<<<dim3(16, 128), dim3(256), smem, stream>>>(Qb, Kb, Vb, attn, Ob);

  outproj_kernel<<<dim3(128, 16), blk, 0, stream>>>(Ob, Wp, bp, query, Ypre);
  ln_kernel<<<dim3(1024), dim3(256), 0, stream>>>(Ypre, gamma, beta, y);
}
